// gnn_dsse_65085934403701
// MI455X (gfx1250) — matrix operand order repaired and checked by reading, not compiled, not run on hardware
//
#include <hip/hip_runtime.h>

#define NN 50000
#define EE 800000
#define CC 128
#define DD 256
#define NLAYERS 5

typedef _Float16 half_t;
typedef __attribute__((ext_vector_type(16))) _Float16 v16h;
typedef __attribute__((ext_vector_type(8)))  float    v8f;
typedef __attribute__((ext_vector_type(4)))  unsigned int v4u;
typedef __attribute__((ext_vector_type(8)))  int      v8i_t;
typedef __attribute__((ext_vector_type(4)))  int      v4i_t;

// ---------------- norm precompute ----------------

__global__ void zero_f32_kernel(float* __restrict__ p, long long n) {
    long long i = (long long)blockIdx.x * blockDim.x + threadIdx.x;
    if (i < n) p[i] = 0.0f;
}

__global__ void degree_kernel(const int* __restrict__ ei, unsigned* __restrict__ deg, int e_cnt) {
    int e = blockIdx.x * blockDim.x + threadIdx.x;
    if (e < e_cnt) atomicAdd(&deg[ei[e_cnt + e]], 1u);
}

__global__ void dinv_kernel(float* __restrict__ d, int n) {
    int i = blockIdx.x * blockDim.x + threadIdx.x;
    if (i < n) {
        unsigned c = ((const unsigned*)d)[i];
        d[i] = rsqrtf((float)c + 1.0f);   // +1 = self loop; always > 0
    }
}

__global__ void norm_kernel(const int* __restrict__ ei, const float* __restrict__ dinv,
                            float* __restrict__ nrm, int e_cnt) {
    int e = blockIdx.x * blockDim.x + threadIdx.x;
    if (e < e_cnt) nrm[e] = dinv[ei[e]] * dinv[ei[e_cnt + e]];
}

// ---------------- weight pre-transpose to f16 ----------------
// wt[l][n][k] = W[l][k][n]  (5 layers of 128x128)
__global__ void prep_wt_kernel(const float* __restrict__ W, half_t* __restrict__ wt) {
    int idx = blockIdx.x * blockDim.x + threadIdx.x;   // 5*128*128
    if (idx >= NLAYERS * CC * CC) return;
    int l = idx >> 14;
    int rem = idx & 16383;
    int n = rem >> 7;
    int k = rem & 127;
    wt[idx] = (half_t)W[l * CC * CC + k * CC + n];     // idx == l*16384 + n*128 + k
}
// wdT[n][k] = Wd[k][n]  (256x128)
__global__ void prep_wdT_kernel(const float* __restrict__ Wd, half_t* __restrict__ wdT) {
    int idx = blockIdx.x * blockDim.x + threadIdx.x;   // 256*128
    if (idx >= DD * CC) return;
    int n = idx >> 7;
    int k = idx & 127;
    wdT[idx] = (half_t)Wd[(size_t)k * DD + n];
}

// ---------------- edge scatter: h[col] += norm * x[row] ----------------

__global__ __launch_bounds__(256) void scatter_kernel(
    const int* __restrict__ ei, const float* __restrict__ nrm,
    const float* __restrict__ x, float* __restrict__ h, int e_cnt)
{
    long long tid = (long long)blockIdx.x * blockDim.x + threadIdx.x;
    int e = (int)(tid >> 5);
    if (e >= e_cnt) return;
    int lane = (int)(tid & 31);
    int r = ei[e];
    int c = ei[e_cnt + e];
    float w = nrm[e];
    const float4 xv = *(const float4*)(x + (size_t)r * CC + lane * 4);
    float* hp = h + (size_t)c * CC + lane * 4;
    unsafeAtomicAdd(hp + 0, w * xv.x);
    unsafeAtomicAdd(hp + 1, w * xv.y);
    unsafeAtomicAdd(hp + 2, w * xv.z);
    unsafeAtomicAdd(hp + 3, w * xv.w);
}

// ---------------- TDM: 2D f16 tile DMA global -> LDS ----------------
// dim0 = row length (halfs, contiguous), dim1 = #rows; tile == tensor.
__device__ __forceinline__ void tdm_load_2d_f16(const half_t* gsrc, unsigned lds_off,
                                                unsigned dim0, unsigned dim1)
{
    unsigned long long ga = (unsigned long long)(size_t)gsrc;
    v4u g0;
    g0.x = 1u;                                             // count=1 valid, user mode
    g0.y = lds_off;                                        // lds_addr
    g0.z = (unsigned)(ga & 0xFFFFFFFFu);                   // global_addr lo
    g0.w = (unsigned)((ga >> 32) & 0x01FFFFFFu)            // global_addr hi (57-bit)
         | 0x80000000u;                                    // type = 2 ("image")
    v8i_t g1;
    g1[0] = (int)(1u << 16);                               // data_size = 1 -> 2 bytes
    g1[1] = (int)(dim0 << 16);                             // tensor_dim0[15:0] @ bits 63:48
    g1[2] = (int)((dim0 >> 16) | (dim1 << 16));            // tensor_dim0 hi | tensor_dim1 lo
    g1[3] = (int)((dim1 >> 16) | (dim0 << 16));            // tensor_dim1 hi | tile_dim0
    g1[4] = (int)(dim1 & 0xFFFFu);                         // tile_dim1 (tile_dim2 = 0)
    g1[5] = (int)dim0;                                     // tensor_dim0_stride lo
    g1[6] = 0;                                             // stride hi / dim1_stride lo
    g1[7] = 0;
    v4i_t z4 = {0, 0, 0, 0};
#if __clang_major__ >= 23
    v8i_t z8 = {0, 0, 0, 0, 0, 0, 0, 0};
    __builtin_amdgcn_tensor_load_to_lds(g0, g1, z4, z4, z8, 0);
#else
    __builtin_amdgcn_tensor_load_to_lds(g0, g1, z4, z4, 0);
#endif
}

// ---------------- WMMA fragment builders (ISA 7.12.2, row stride 128 halfs) ----------------
__device__ __forceinline__ v16h load_a_frag(const half_t* __restrict__ tile, int lane, int k0) {
    int r  = lane & 15;
    int hi = lane >> 4;
    v16h f;
#pragma unroll
    for (int j = 0; j < 8; ++j) {
        int k = k0 + ((j < 4) ? (hi * 8 + 2 * j) : (16 + hi * 8 + 2 * (j - 4)));
        f[2 * j]     = tile[r * CC + k];
        f[2 * j + 1] = tile[r * CC + k + 1];
    }
    return f;
}
__device__ __forceinline__ v16h load_b_frag(const half_t* __restrict__ tile, int lane, int k0) {
    int n  = lane & 15;
    int hi = lane >> 4;
    v16h f;
#pragma unroll
    for (int j = 0; j < 8; ++j) {
        int k = k0 + ((j < 4) ? (hi * 8 + 2 * j) : (16 + hi * 8 + 2 * (j - 4)));
        f[2 * j]     = tile[n * CC + k];
        f[2 * j + 1] = tile[n * CC + k + 1];
    }
    return f;
}

// ---------------- fused combine + 128x128 GEMM + leaky-relu ----------------
// a = 0.9*(h + dinv^2 * x) + 0.1*x0 ; xnext = leaky_relu(a @ W)
// block = 128 thr (4 waves); block tile = 64 rows x 128 cols; full K staged once.

__global__ __launch_bounds__(128) void gcn_layer_kernel(
    const float* __restrict__ h, const float* __restrict__ xcur,
    const float* __restrict__ x0, const float* __restrict__ dinv,
    const half_t* __restrict__ wtL, float* __restrict__ xnext)
{
    __shared__ half_t aT[64 * CC];    // 16 KB
    __shared__ half_t bT[128 * CC];   // 32 KB
    const int tid  = threadIdx.x;
    const int wave = tid >> 5, lane = tid & 31;
    const int rowBase = blockIdx.x * 64;

    // TDM: pull whole 128x128 f16 weight panel into LDS (one DMA per block)
    if (tid == 0)
        tdm_load_2d_f16(wtL, (unsigned)(size_t)bT, CC, 128u);

    // stage A (64 x 128) with fused combine; thread covers half a row
    {
        int r  = tid >> 1;
        int cb = (tid & 1) * 64;
        int gr = rowBase + r;
        if (gr < NN) {
            float dv  = dinv[gr];
            float dv2 = dv * dv;
            const float* hp  = h    + (size_t)gr * CC + cb;
            const float* xp  = xcur + (size_t)gr * CC + cb;
            const float* x0p = x0   + (size_t)gr * CC + cb;
#pragma unroll
            for (int c = 0; c < 64; ++c) {
                float a = 0.9f * (hp[c] + dv2 * xp[c]) + 0.1f * x0p[c];
                aT[r * CC + cb + c] = (half_t)a;
            }
        } else {
#pragma unroll
            for (int c = 0; c < 64; ++c) aT[r * CC + cb + c] = (half_t)0.0f;
        }
    }
    __builtin_amdgcn_s_wait_tensorcnt(0);
    __syncthreads();

    v8f acc[8];
#pragma unroll
    for (int t = 0; t < 8; ++t)
#pragma unroll
        for (int i = 0; i < 8; ++i) acc[t][i] = 0.0f;

    v16h af[4];
#pragma unroll
    for (int kc = 0; kc < 4; ++kc)
        af[kc] = load_a_frag(&aT[wave * 16 * CC], lane, kc * 32);

#pragma unroll
    for (int t = 0; t < 8; ++t) {
#pragma unroll
        for (int kc = 0; kc < 4; ++kc) {
            v16h bf = load_b_frag(&bT[t * 16 * CC], lane, kc * 32);
            acc[t] = __builtin_amdgcn_wmma_f32_16x16x32_f16(
                false, af[kc], false, bf, (short)0, acc[t], false, false);
        }
    }

    // epilogue: leaky relu (C/D layout: vgpr i -> M=i(+8 hi lanes), N=lane&15)
    const int mOff = (lane >> 4) * 8;
    const int n0   = lane & 15;
#pragma unroll
    for (int t = 0; t < 8; ++t)
#pragma unroll
        for (int i = 0; i < 8; ++i) {
            int gr = rowBase + wave * 16 + mOff + i;
            if (gr < NN) {
                float v = acc[t][i];
                v = (v > 0.0f) ? v : 0.01f * v;
                xnext[(size_t)gr * CC + t * 16 + n0] = v;
            }
        }
}

// ---------------- dense: t = x @ Wd (128x256) + bd ----------------
// grid.y in {0,1} selects N-half; same 64x128 structure as layer kernel.

__global__ __launch_bounds__(128) void dense_kernel(
    const float* __restrict__ x, const half_t* __restrict__ wdT,
    const float* __restrict__ bd, float* __restrict__ tout)
{
    __shared__ half_t aT[64 * CC];    // 16 KB
    __shared__ half_t bT[128 * CC];   // 32 KB
    const int tid  = threadIdx.x;
    const int wave = tid >> 5, lane = tid & 31;
    const int rowBase = blockIdx.x * 64;
    const int nBase   = blockIdx.y * 128;

    if (tid == 0)
        tdm_load_2d_f16(wdT + (size_t)nBase * CC, (unsigned)(size_t)bT, CC, 128u);

    {
        int r  = tid >> 1;
        int cb = (tid & 1) * 64;
        int gr = rowBase + r;
        if (gr < NN) {
            const float* xp = x + (size_t)gr * CC + cb;
#pragma unroll
            for (int c = 0; c < 64; ++c) aT[r * CC + cb + c] = (half_t)xp[c];
        } else {
#pragma unroll
            for (int c = 0; c < 64; ++c) aT[r * CC + cb + c] = (half_t)0.0f;
        }
    }
    __builtin_amdgcn_s_wait_tensorcnt(0);
    __syncthreads();

    v8f acc[8];
#pragma unroll
    for (int t = 0; t < 8; ++t)
#pragma unroll
        for (int i = 0; i < 8; ++i) acc[t][i] = 0.0f;

    v16h af[4];
#pragma unroll
    for (int kc = 0; kc < 4; ++kc)
        af[kc] = load_a_frag(&aT[wave * 16 * CC], lane, kc * 32);

#pragma unroll
    for (int t = 0; t < 8; ++t) {
#pragma unroll
        for (int kc = 0; kc < 4; ++kc) {
            v16h bf = load_b_frag(&bT[t * 16 * CC], lane, kc * 32);
            acc[t] = __builtin_amdgcn_wmma_f32_16x16x32_f16(
                false, af[kc], false, bf, (short)0, acc[t], false, false);
        }
    }

    const int mOff = (lane >> 4) * 8;
    const int n0   = lane & 15;
#pragma unroll
    for (int t = 0; t < 8; ++t)
#pragma unroll
        for (int i = 0; i < 8; ++i) {
            int gr = rowBase + wave * 16 + mOff + i;
            int n  = nBase + t * 16 + n0;
            if (gr < NN)
                tout[(size_t)gr * DD + n] = acc[t][i] + bd[n];
        }
}

// ---------------- output head: out = t @ Wo (256x2) + bo ----------------

__global__ __launch_bounds__(128) void out_kernel(
    const float* __restrict__ t, const float* __restrict__ Wo,
    const float* __restrict__ bo, float* __restrict__ out)
{
    int n = blockIdx.x * blockDim.x + threadIdx.x;
    if (n >= NN) return;
    float s0 = bo[0], s1 = bo[1];
    const float* tp = t + (size_t)n * DD;
#pragma unroll 4
    for (int k = 0; k < DD; ++k) {
        float v = tp[k];
        s0 += v * Wo[k * 2 + 0];
        s1 += v * Wo[k * 2 + 1];
    }
    out[n * 2 + 0] = s0;
    out[n * 2 + 1] = s1;
}

// ---------------- launch ----------------

extern "C" void kernel_launch(void* const* d_in, const int* in_sizes, int n_in,
                              void* d_out, int out_size, void* d_ws, size_t ws_size,
                              hipStream_t stream) {
    const float* x_in  = (const float*)d_in[0];
    const int*   ei    = (const int*)d_in[1];
    const float* convW = (const float*)d_in[2];
    const float* wd    = (const float*)d_in[3];
    const float* bd    = (const float*)d_in[4];
    const float* wo    = (const float*)d_in[5];
    const float* bo    = (const float*)d_in[6];
    float*       out   = (float*)d_out;

    // workspace layout (floats); tbuf (NN*DD) aliases h+xb which are dead by then
    float* ws   = (float*)d_ws;
    float* dinv = ws;                 // NN (used as u32 degree first)
    float* nrm  = ws + 50048;         // EE
    float* xa   = ws + 850048;        // NN*CC
    float* h    = ws + 7250048;       // NN*CC
    float* xb   = ws + 13650048;      // NN*CC
    float* tbuf = h;                  // NN*DD == (h + xb) span
    half_t* hb  = (half_t*)(ws + 20050048);
    half_t* wt5 = hb;                 // NLAYERS * CC * CC halfs (pre-transposed f16)
    half_t* wdT = hb + NLAYERS * CC * CC;  // DD * CC halfs

    // gcn_norm
    zero_f32_kernel<<<(NN + 255) / 256, 256, 0, stream>>>(dinv, NN);
    degree_kernel<<<(EE + 255) / 256, 256, 0, stream>>>(ei, (unsigned*)dinv, EE);
    dinv_kernel<<<(NN + 255) / 256, 256, 0, stream>>>(dinv, NN);
    norm_kernel<<<(EE + 255) / 256, 256, 0, stream>>>(ei, dinv, nrm, EE);

    // weight pre-transpose/convert (f32 -> f16, n-major)
    prep_wt_kernel<<<(NLAYERS * CC * CC + 255) / 256, 256, 0, stream>>>(convW, wt5);
    prep_wdT_kernel<<<(DD * CC + 255) / 256, 256, 0, stream>>>(wd, wdT);

    const long long hElems = (long long)NN * CC;
    const int scatterBlocks = EE / 8;           // EE*32 threads / 256
    const int gemmBlocks = (NN + 63) / 64;

    const float* xcur = x_in;
    float* bufs[2] = { xa, xb };
    for (int l = 0; l < NLAYERS; ++l) {
        float* xnext = bufs[l & 1];
        zero_f32_kernel<<<(unsigned)((hElems + 255) / 256), 256, 0, stream>>>(h, hElems);
        scatter_kernel<<<scatterBlocks, 256, 0, stream>>>(ei, nrm, xcur, h, EE);
        gcn_layer_kernel<<<gemmBlocks, 128, 0, stream>>>(
            h, xcur, x_in, dinv, wt5 + (size_t)l * CC * CC, xnext);
        xcur = xnext;
    }

    dense_kernel<<<dim3(gemmBlocks, 2), 128, 0, stream>>>(xcur, wdT, bd, tbuf);
    out_kernel<<<(NN + 127) / 128, 128, 0, stream>>>(tbuf, wo, bo, out);
}
